// DynamicConv2d_66864050864932
// MI455X (gfx1250) — matrix operand order repaired and checked by reading, not compiled, not run on hardware
//
#include <hip/hip_runtime.h>
#include <hip/hip_bf16.h>

// ---- CDNA5 WMMA vector types -------------------------------------------------
typedef __bf16 v16bf __attribute__((ext_vector_type(16)));
typedef __bf16 v8bf  __attribute__((ext_vector_type(8)));
typedef float  v8f   __attribute__((ext_vector_type(8)));

typedef unsigned int u32x4 __attribute__((ext_vector_type(4)));
typedef int          i32x8 __attribute__((ext_vector_type(8)));
typedef int          i32x4 __attribute__((ext_vector_type(4)));

union V16 { v16bf v; v8bf h[2]; };

// Problem constants
#define BATCH 16
#define CIN   64
#define COUT  64
#define HIN   192
#define WIN   192
#define HOUT  190
#define WOUT  190
#define DEG   64
#define HID   256
#define KDIM  576              // CIN*3*3, conv GEMM reduction length
#define NW2   36864            // COUT*CIN*3*3

// Conv tiling
#define GY    5                // output rows per workgroup (190 = 38*5)
#define XROWS (GY + 2)         // 7 input rows staged in LDS
#define WROW  584              // padded LDS K-stride for weight rows (1168B: lane stride 36 dw mod 64 -> conflict-free)
#define CSTR  72               // padded LDS channel stride for x tile (144B: same property)

#define SMEM_W_BYTES (COUT * WROW * 2)               // 74752
#define SMEM_X_BYTES (XROWS * 34 * CSTR * 2)         // 34272
#define SMEM_BYTES   (SMEM_W_BYTES + SMEM_X_BYTES)   // 109024 (< 320KB WGP LDS)

#if defined(__has_builtin)
# if __has_builtin(__builtin_amdgcn_tensor_load_to_lds) && __has_builtin(__builtin_amdgcn_s_wait_tensorcnt)
#  define USE_TDM 1
# endif
#endif
#ifndef USE_TDM
# define USE_TDM 0
#endif

// ============================================================================
// Stage 1: h = relu(deg @ W1 + b1) -> bf16 [16, 256] in workspace
// ============================================================================
__global__ void gen_h_kernel(const float* __restrict__ deg,
                             const float* __restrict__ W1,
                             const float* __restrict__ b1,
                             __bf16* __restrict__ hbf) {
  int b = blockIdx.x, j = threadIdx.x;
  float s = b1[j];
  #pragma unroll 8
  for (int k = 0; k < DEG; ++k)
    s += deg[b * DEG + k] * W1[k * HID + j];
  hbf[b * HID + j] = (__bf16)fmaxf(s, 0.0f);
}

// ============================================================================
// Stage 2: weights = h @ W2 + b2 via bf16 WMMA (M=16 samples, K=256, N=36864),
// scattered to per-sample transposed bf16 layout wt[b][cout][(kh*3+kw)*64+cin].
// ============================================================================
__global__ void __launch_bounds__(256)
wgen_wmma_kernel(const __bf16* __restrict__ hbf,
                 const float* __restrict__ W2,
                 const float* __restrict__ b2,
                 __bf16* __restrict__ wt) {
  int wave = (int)((blockIdx.x * blockDim.x + threadIdx.x) >> 5);
  int lane = threadIdx.x & 31;
  int n0   = wave * 16;                       // 2304 tiles = 288 blocks * 8 waves
  int lrow = lane & 15;
  bool hi  = lane >= 16;
  int col  = n0 + lrow;

  v8f acc = {};
  #pragma unroll
  for (int c = 0; c < 8; ++c) {               // K = 256 in chunks of 32
    int k0 = c * 32;
    V16 a;                                    // A layout: lo lanes K0-7/K16-23, hi K8-15/K24-31
    int aoff = k0 + (hi ? 8 : 0);
    a.h[0] = *(const v8bf*)(hbf + lrow * HID + aoff);
    a.h[1] = *(const v8bf*)(hbf + lrow * HID + aoff + 16);
    V16 bf;                                   // B layout: lo lanes K0-7/K16-23, hi K8-15/K24-31
    int koff = k0 + (hi ? 8 : 0);
    #pragma unroll
    for (int e = 0; e < 16; ++e)
      bf.v[e] = (__bf16)W2[(size_t)(koff + e + (e >= 8 ? 8 : 0)) * NW2 + col];
    acc = __builtin_amdgcn_wmma_f32_16x16x32_bf16(
        false, a.v, false, bf.v, (short)0, acc, false, false);
  }
  #pragma unroll
  for (int v = 0; v < 8; ++v) {
    int m    = hi ? v + 8 : v;                // sample (C/D VGPR layout)
    float val = acc[v] + b2[col];
    int cout = col / KDIM;
    int rem  = col - cout * KDIM;
    int cin  = rem / 9;
    int tap  = rem - cin * 9;
    wt[(size_t)(m * COUT + cout) * KDIM + tap * CIN + cin] = (__bf16)val;
  }
}

// ============================================================================
// Stage 3: per-sample implicit-GEMM conv, bf16 WMMA.
// Workgroup = (b, 5 output rows, 32 output cols). 8 waves: 4 cout-tiles x 2 w-subtiles.
// Weights DMA'd to LDS by the Tensor Data Mover (iterate mode adds the row pad),
// x tile filled by all threads concurrently.
// ============================================================================
__global__ void __launch_bounds__(256)
dynconv_wmma_kernel(const float* __restrict__ x,
                    const __bf16* __restrict__ wt,
                    float* __restrict__ out) {
  extern __shared__ char smem[];
  __bf16* wsh = (__bf16*)smem;                       // [COUT][WROW] (rows padded 576->584)
  __bf16* xs  = (__bf16*)(smem + SMEM_W_BYTES);      // [7][34][CSTR] ch-innermost

  int tid = threadIdx.x;
  int b   = blockIdx.z;
  int h0  = blockIdx.y * GY;
  int w0  = blockIdx.x * 32;

  const __bf16* wsrc = wt + (size_t)b * COUT * KDIM;

#if USE_TDM
  // ---- TDM: DMA 64 rows of 576 bf16 into LDS with stride 584 (iterate mode) ----
  if (tid < 32) {                       // wave 0 issues; EXEC ignored by tensor ops
    unsigned long long ga = (unsigned long long)(uintptr_t)wsrc;
    u32x4 g0; i32x8 g1; i32x4 g2, g3; i32x8 g4;
    g0[0] = 1u;                                       // count=1, no gather
    g0[1] = 0u;                                       // lds_addr = 0 (wsh at LDS base)
    g0[2] = (unsigned)(ga & 0xffffffffu);             // global_addr[31:0]
    g0[3] = ((unsigned)(ga >> 32) & 0x01ffffffu) | 0x80000000u; // addr[56:32] | type=2
    g1[0] = (1 << 16) | (1 << 19);                    // data_size=2B, iterate_enable
    g1[1] = (int)((KDIM & 0xffff) << 16);             // tensor_dim0 = 576 (low16)
    g1[2] = (1 << 16);                                // tensor_dim0 hi=0, tensor_dim1 = 1
    g1[3] = (int)((KDIM & 0xffff) << 16);             // tile_dim0 = 576
    g1[4] = 0;                                        // tile_dim1/2 unused
    g1[5] = KDIM;                                     // tensor_dim0_stride = 576
    g1[6] = 0; g1[7] = 0;
    g2[0] = 0;                                        // tensor_dim2 (unused)
    g2[1] = WROW;                                     // lds_addr_increment = 584 elems
    g2[2] = KDIM;                                     // global_addr_increment = 576 elems
    g2[3] = (63 << 16);                               // iterate_count = 63 -> 64 rows
    g3[0] = 0; g3[1] = 0; g3[2] = 0; g3[3] = 0;
    g4[0] = 0; g4[1] = 0; g4[2] = 0; g4[3] = 0;
    g4[4] = 0; g4[5] = 0; g4[6] = 0; g4[7] = 0;
    // 6-arg form on this toolchain: (g0, g1, g2, g3, extra_i32x8, cpol)
    __builtin_amdgcn_tensor_load_to_lds(g0, g1, g2, g3, g4, 0);
  }
#else
  // ---- fallback: manual copy, 4 threads per weight row, padded dst stride ----
  {
    int row = tid >> 2, part = tid & 3;
    const uint4* s = (const uint4*)((const char*)wsrc + row * (KDIM * 2) + part * 288);
    uint4* d = (uint4*)((char*)wsh + row * (WROW * 2) + part * 288);
    #pragma unroll
    for (int i = 0; i < 18; ++i) d[i] = s[i];
  }
#endif

  // ---- fill x tile: rows h0..h0+6, cols w0..w0+33, 64 channels, f32->bf16 ----
  // Main 32 cols (shift-only index math, coalesced 32-element runs per (r,c)):
  for (int e = tid; e < XROWS * CIN * 32; e += 256) {
    int r   = e >> 11;                 // /(64*32)
    int rem = e & 2047;
    int c   = rem >> 5;
    int j   = rem & 31;
    xs[(r * 34 + j) * CSTR + c] =
        (__bf16)x[((size_t)(b * CIN + c) * HIN + (h0 + r)) * WIN + (w0 + j)];
  }
  // Cols 32,33 (clamped at the right edge; contributes only to masked outputs):
  for (int e = tid; e < XROWS * CIN * 2; e += 256) {
    int r   = e >> 7;                  // /(64*2)
    int rem = e & 127;
    int c   = rem >> 1;
    int j   = 32 + (rem & 1);
    int wc  = w0 + j; if (wc > WIN - 1) wc = WIN - 1;
    xs[(r * 34 + j) * CSTR + c] =
        (__bf16)x[((size_t)(b * CIN + c) * HIN + (h0 + r)) * WIN + wc];
  }

#if USE_TDM
  if (tid < 32) __builtin_amdgcn_s_wait_tensorcnt(0);
#endif
  __syncthreads();

  int wave = tid >> 5, lane = tid & 31;
  int mt   = wave & 3;                 // cout tile (16 couts)
  int sub  = wave >> 2;                // w subtile (16 cols)
  int lrow = lane & 15;
  bool hi  = lane >= 16;

  const __bf16* abase = wsh + (mt * 16 + lrow) * WROW + (hi ? 8 : 0);
  int boff0 = hi ? 8 : 0;

  v8f acc[GY] = {};
  #pragma unroll
  for (int tap = 0; tap < 9; ++tap) {
    int kh = tap / 3, kw = tap - kh * 3;
    #pragma unroll
    for (int cc = 0; cc < CIN; cc += 32) {
      V16 a;                           // one A fragment reused across GY rows
      const __bf16* ap = abase + tap * CIN + cc;
      a.h[0] = *(const v8bf*)(ap);
      a.h[1] = *(const v8bf*)(ap + 16);
      #pragma unroll
      for (int r = 0; r < GY; ++r) {
        const __bf16* bp = xs + ((kh + r) * 34 + sub * 16 + lrow + kw) * CSTR + cc + boff0;
        V16 bf;
        bf.h[0] = *(const v8bf*)(bp);
        bf.h[1] = *(const v8bf*)(bp + 16);
        acc[r] = __builtin_amdgcn_wmma_f32_16x16x32_bf16(
            false, a.v, false, bf.v, (short)0, acc[r], false, false);
      }
    }
  }

  // ---- store GY x (16x16) f32 tiles, mask out-of-range columns ----
  int wout = w0 + sub * 16 + lrow;
  if (wout < WOUT) {
    #pragma unroll
    for (int r = 0; r < GY; ++r) {
      #pragma unroll
      for (int v = 0; v < 8; ++v) {
        int cout = mt * 16 + (hi ? v + 8 : v);
        out[((size_t)(b * COUT + cout) * HOUT + (h0 + r)) * WOUT + wout] = acc[r][v];
      }
    }
  }
}

// ============================================================================
extern "C" void kernel_launch(void* const* d_in, const int* in_sizes, int n_in,
                              void* d_out, int out_size, void* d_ws, size_t ws_size,
                              hipStream_t stream) {
  (void)in_sizes; (void)n_in; (void)out_size; (void)ws_size;
  const float* x   = (const float*)d_in[0];
  const float* deg = (const float*)d_in[1];
  const float* W1  = (const float*)d_in[2];
  const float* b1  = (const float*)d_in[3];
  const float* W2  = (const float*)d_in[4];
  const float* b2  = (const float*)d_in[5];
  float* out = (float*)d_out;

  __bf16* hbf = (__bf16*)d_ws;                 // [16*256] bf16
  __bf16* wtr = hbf + BATCH * HID;             // [16*64*576] bf16 transposed weights

  gen_h_kernel<<<BATCH, HID, 0, stream>>>(deg, W1, b1, hbf);
  wgen_wmma_kernel<<<288, 256, 0, stream>>>(hbf, W2, b2, wtr);
  dim3 grid(6, HOUT / GY, BATCH);              // 6 w-tiles x 38 row-groups x 16 samples
  dynconv_wmma_kernel<<<grid, 256, SMEM_BYTES, stream>>>(x, wtr, out);
}
